// RNNLM_661424963826
// MI455X (gfx1250) — compile-verified, matrix operand order checked by reading, not run on hardware
//
#include <hip/hip_runtime.h>
#include <stdint.h>

#define T_LEN 128
#define B_SZ 128
#define N_NOISE 1000
#define E_DIM 1024
#define H_DIM 1024
#define G4H 4096
#define NTILES_NOISE 63   // ceil(1000/16)

typedef __attribute__((ext_vector_type(16))) __bf16 v16bf;
typedef __attribute__((ext_vector_type(8)))  float  v8f;

union BF16Tile {
  v16bf v;
  uint4 q[2];
  uint32_t u[8];
};

__device__ __forceinline__ uint16_t f2bf(float x) {
  uint32_t u = __float_as_uint(x);
  u += 0x7fffu + ((u >> 16) & 1u);          // round-to-nearest-even
  return (uint16_t)(u >> 16);
}
__device__ __forceinline__ float bf2f(uint16_t h) {
  return __uint_as_float(((uint32_t)h) << 16);
}
__device__ __forceinline__ uint32_t pk2(float lo, float hi) {
  return (uint32_t)f2bf(lo) | ((uint32_t)f2bf(hi) << 16);
}
__device__ __forceinline__ float sigm(float x) { return 1.0f / (1.0f + __expf(-x)); }

__device__ __forceinline__ v8f wmma_bf16(const BF16Tile& a, const BF16Tile& b, v8f c) {
  return __builtin_amdgcn_wmma_f32_16x16x32_bf16(false, a.v, false, b.v,
                                                 (short)0, c, false, false);
}

// ---------------- prep kernels ----------------

__global__ void cvt_weights(const float* __restrict__ wih, const float* __restrict__ whh,
                            uint16_t* __restrict__ wihb, uint16_t* __restrict__ whhb) {
  size_t i = (size_t)blockIdx.x * 256 + threadIdx.x;   // 4,194,304 each
  if (i < (size_t)G4H * E_DIM) {
    wihb[i] = f2bf(wih[i]);
    whhb[i] = f2bf(whh[i]);
  }
}

__global__ void prep_state(const float* __restrict__ bih, const float* __restrict__ bhh,
                           float* __restrict__ bias_sum,
                           const float* __restrict__ h0, const float* __restrict__ c0,
                           uint16_t* __restrict__ h_bf, float* __restrict__ cbuf) {
  int i = blockIdx.x * 256 + threadIdx.x;              // 131072 threads
  if (i < G4H) bias_sum[i] = bih[i] + bhh[i];
  if (i < B_SZ * H_DIM) {
    h_bf[i] = f2bf(h0[i]);
    cbuf[i] = c0[i];
  }
}

__global__ void emb_gather(const int* __restrict__ ids, const float* __restrict__ tab,
                           uint16_t* __restrict__ emb_bf) {
  size_t i = (size_t)blockIdx.x * 256 + threadIdx.x;   // 16,777,216
  if (i >= (size_t)T_LEN * B_SZ * E_DIM) return;
  size_t tb = i >> 10;
  int k = (int)(i & 1023);
  int id = ids[tb];
  emb_bf[i] = f2bf(tab[(size_t)id * E_DIM + k]);
}

// ---------------- LSTM step: gate GEMM ----------------
// gates[b, j] = sum_k x[b,k]*Wih[j,k] + sum_k h[b,k]*Whh[j,k] + bias_sum[j]
// Each wave computes a 16x64 strip (4 WMMA accumulators) so the A tile is
// loaded once per 4 WMMAs. grid (8, 8), block 256 (8 waves).
__global__ void lstm_gates(const uint16_t* __restrict__ emb_bf,   // [T*B, E]
                           const uint16_t* __restrict__ h_bf,     // [B, H]
                           const uint16_t* __restrict__ wihb,     // [4H, E]
                           const uint16_t* __restrict__ whhb,     // [4H, H]
                           const float* __restrict__ bias_sum,    // [4H]
                           float* __restrict__ gates,             // [B, 4H]
                           int t) {
  const int wave = threadIdx.x >> 5;
  const int lane = threadIdx.x & 31;
  const int nquad = blockIdx.x * 8 + wave;     // 0..63 -> 64 output cols each
  const int mtile = blockIdx.y;                // 0..7
  const int n0 = nquad * 64, m0 = mtile * 16;
  const int half = lane >> 4, lm = lane & 15;

  const uint16_t* arow_x = emb_bf + ((size_t)t * B_SZ + (m0 + lm)) * E_DIM;
  const uint16_t* arow_h = h_bf + (size_t)(m0 + lm) * H_DIM;

  v8f acc[4] = {};

  {
    const uint16_t* bp[4];
#pragma unroll
    for (int q = 0; q < 4; ++q) bp[q] = wihb + (size_t)(n0 + q * 16 + lm) * E_DIM;
#pragma unroll 2
    for (int k0 = 0; k0 < E_DIM; k0 += 32) {
      const int kc = k0 + half * 16;
      BF16Tile a;
      a.q[0] = *(const uint4*)(arow_x + kc);
      a.q[1] = *(const uint4*)(arow_x + kc + 8);
#pragma unroll
      for (int q = 0; q < 4; ++q) {
        BF16Tile b;
        b.q[0] = *(const uint4*)(bp[q] + kc);
        b.q[1] = *(const uint4*)(bp[q] + kc + 8);
        acc[q] = wmma_bf16(a, b, acc[q]);
      }
    }
  }
  {
    const uint16_t* bp[4];
#pragma unroll
    for (int q = 0; q < 4; ++q) bp[q] = whhb + (size_t)(n0 + q * 16 + lm) * H_DIM;
#pragma unroll 2
    for (int k0 = 0; k0 < H_DIM; k0 += 32) {
      const int kc = k0 + half * 16;
      BF16Tile a;
      a.q[0] = *(const uint4*)(arow_h + kc);
      a.q[1] = *(const uint4*)(arow_h + kc + 8);
#pragma unroll
      for (int q = 0; q < 4; ++q) {
        BF16Tile b;
        b.q[0] = *(const uint4*)(bp[q] + kc);
        b.q[1] = *(const uint4*)(bp[q] + kc + 8);
        acc[q] = wmma_bf16(a, b, acc[q]);
      }
    }
  }
#pragma unroll
  for (int q = 0; q < 4; ++q) {
    const int col = n0 + q * 16 + lm;
    const float bs = bias_sum[col];
#pragma unroll
    for (int r = 0; r < 8; ++r) {
      const int row = m0 + r + half * 8;
      gates[(size_t)row * G4H + col] = acc[q][r] + bs;
    }
  }
}

// ---------------- LSTM step: cell update ----------------
__global__ void lstm_cell(const float* __restrict__ gates, float* __restrict__ cbuf,
                          uint16_t* __restrict__ h_bf, uint16_t* __restrict__ hs_bf,
                          const int* __restrict__ sent_lens, int t) {
  int idx = blockIdx.x * blockDim.x + threadIdx.x;     // 0 .. 131071
  if (idx >= B_SZ * H_DIM) return;
  const int b = idx >> 10, j = idx & 1023;
  const float* g = gates + (size_t)b * G4H;
  const float ig = sigm(g[j]);
  const float fg = sigm(g[H_DIM + j]);
  const float gg = tanhf(g[2 * H_DIM + j]);
  const float og = sigm(g[3 * H_DIM + j]);
  const float c = fg * cbuf[idx] + ig * gg;
  const float h = og * tanhf(c);
  cbuf[idx] = c;
  const uint16_t hb = f2bf(h);
  h_bf[idx] = hb;
  hs_bf[(size_t)t * (B_SZ * H_DIM) + idx] = (t < sent_lens[b]) ? hb : (uint16_t)0;
}

// ---------------- NCE noise GEMM + fused loss epilogue ----------------
// out[t,b,n] = hs[t,b,:] . nce_w[noise[t,n],:] + nce_b[...]
// Each wave owns ONE n-tile and iterates over all 8 m-tiles, so the gathered
// f32->bf16 B tile is converted once and reused by 8 WMMAs.
// grid (8, 128), block 256 (8 waves -> 8 consecutive n-tiles).
__global__ void nce_noise(const uint16_t* __restrict__ hs_bf, const float* __restrict__ nce_w,
                          const float* __restrict__ nce_b, const float* __restrict__ noiseprob,
                          const int* __restrict__ noise, float* __restrict__ NLpart) {
  const int wave = threadIdx.x >> 5;
  const int lane = threadIdx.x & 31;
  const int ntile = blockIdx.x * 8 + wave;     // 0..63
  const int t = blockIdx.y;                    // 0..127
  if (ntile >= NTILES_NOISE) return;           // wave-uniform exit
  const int n0 = ntile * 16;
  const int half = lane >> 4, lm = lane & 15;

  const int ncol = n0 + lm;
  const bool nvalid = (ncol < N_NOISE);
  const int vrow = nvalid ? noise[(size_t)t * N_NOISE + ncol] : 0;
  const float* brow = nce_w + (size_t)vrow * H_DIM;
  const uint16_t* abase = hs_bf + ((size_t)t * B_SZ + lm) * H_DIM;

  v8f acc[8] = {};
  for (int k0 = 0; k0 < H_DIM; k0 += 32) {
    const int kc = k0 + half * 16;
    // stream the gathered f32 weight row (HBM-bound): prefetch ahead
    __builtin_prefetch(brow + kc + 128, 0, 1);
    float4 f0 = *(const float4*)(brow + kc);
    float4 f1 = *(const float4*)(brow + kc + 4);
    float4 f2 = *(const float4*)(brow + kc + 8);
    float4 f3 = *(const float4*)(brow + kc + 12);
    BF16Tile b;
    b.u[0] = pk2(f0.x, f0.y); b.u[1] = pk2(f0.z, f0.w);
    b.u[2] = pk2(f1.x, f1.y); b.u[3] = pk2(f1.z, f1.w);
    b.u[4] = pk2(f2.x, f2.y); b.u[5] = pk2(f2.z, f2.w);
    b.u[6] = pk2(f3.x, f3.y); b.u[7] = pk2(f3.z, f3.w);
#pragma unroll
    for (int m = 0; m < 8; ++m) {
      const uint16_t* arow = abase + (size_t)(m * 16) * H_DIM;
      BF16Tile a;
      a.q[0] = *(const uint4*)(arow + kc);
      a.q[1] = *(const uint4*)(arow + kc + 8);
      acc[m] = wmma_bf16(a, b, acc[m]);
    }
  }
  const float bias = nvalid ? nce_b[vrow] : 0.0f;
  const float Kp = nvalid ? (1000.0f * noiseprob[vrow]) : 0.0f;
#pragma unroll
  for (int m = 0; m < 8; ++m) {
#pragma unroll
    for (int r = 0; r < 8; ++r) {
      const float out = acc[m][r] + bias;
      const float score = __expf(out - 9.0f);
      float contrib = nvalid ? __logf(1e-10f + Kp / (score + Kp)) : 0.0f;
      // sum over the 16 columns (lanes within each half)
      contrib += __shfl_xor(contrib, 1, 32);
      contrib += __shfl_xor(contrib, 2, 32);
      contrib += __shfl_xor(contrib, 4, 32);
      contrib += __shfl_xor(contrib, 8, 32);
      if (lm == 0) {
        const int row = m * 16 + r + half * 8;
        NLpart[((size_t)t * B_SZ + row) * NTILES_NOISE + ntile] = contrib;
      }
    }
  }
}

// ---------------- target diagonal scores ----------------
// ts[t,b] = hs[t,b,:] . nce_w[target[t,b],:] + nce_b[target]
__global__ void nce_diag(const uint16_t* __restrict__ hs_bf, const float* __restrict__ nce_w,
                         const float* __restrict__ nce_b, const int* __restrict__ target,
                         float* __restrict__ ts) {
  const int wid = (blockIdx.x * blockDim.x + threadIdx.x) >> 5;   // one wave per (t,b)
  const int lane = threadIdx.x & 31;
  if (wid >= T_LEN * B_SZ) return;
  const int tg = target[wid];
  const uint16_t* hrow = hs_bf + (size_t)wid * H_DIM;
  const float* wrow = nce_w + (size_t)tg * H_DIM;
  float s = 0.0f;
#pragma unroll
  for (int i = 0; i < 8; ++i) {
    const int k = (lane + i * 32) * 4;
    float4 wv = *(const float4*)(wrow + k);
    ushort4 hv = *(const ushort4*)(hrow + k);
    s += bf2f(hv.x) * wv.x + bf2f(hv.y) * wv.y + bf2f(hv.z) * wv.z + bf2f(hv.w) * wv.w;
  }
  s += __shfl_xor(s, 1, 32);
  s += __shfl_xor(s, 2, 32);
  s += __shfl_xor(s, 4, 32);
  s += __shfl_xor(s, 8, 32);
  s += __shfl_xor(s, 16, 32);
  if (lane == 0) ts[wid] = s + nce_b[tg];
}

// ---------------- final masked reduction ----------------
__global__ void final_reduce(const float* __restrict__ ts, const float* __restrict__ NLpart,
                             const int* __restrict__ target, const float* __restrict__ noiseprob,
                             const int* __restrict__ input_ids, float* __restrict__ out) {
  __shared__ float ssum[256];
  __shared__ float scnt[256];
  float sum = 0.0f, cnt = 0.0f;
  for (int i = threadIdx.x; i < T_LEN * B_SZ; i += 256) {
    const int t = i >> 7;
    float nl = 0.0f;
    const float* p = NLpart + (size_t)i * NTILES_NOISE;
    for (int j = 0; j < NTILES_NOISE; ++j) nl += p[j];
    const int tg = target[i];
    const float score = __expf(ts[i] - 9.0f);
    const float Kp = 1000.0f * noiseprob[tg];
    const float ml = __logf(1e-10f + score / (score + Kp));
    const float loss = -(ml + nl);
    const bool m = (input_ids[i] > 0) || (t == 0);
    if (m) { sum += loss; cnt += 1.0f; }
  }
  ssum[threadIdx.x] = sum;
  scnt[threadIdx.x] = cnt;
  __syncthreads();
  for (int s = 128; s > 0; s >>= 1) {
    if (threadIdx.x < s) {
      ssum[threadIdx.x] += ssum[threadIdx.x + s];
      scnt[threadIdx.x] += scnt[threadIdx.x + s];
    }
    __syncthreads();
  }
  if (threadIdx.x == 0) out[0] = ssum[0] / scnt[0];
}

// ---------------- launcher ----------------
extern "C" void kernel_launch(void* const* d_in, const int* in_sizes, int n_in,
                              void* d_out, int out_size, void* d_ws, size_t ws_size,
                              hipStream_t stream) {
  (void)in_sizes; (void)n_in; (void)out_size; (void)ws_size;
  const int*   input_ids = (const int*)d_in[0];
  const int*   target    = (const int*)d_in[1];
  const int*   noise     = (const int*)d_in[2];
  const int*   sent_lens = (const int*)d_in[3];
  const float* h0        = (const float*)d_in[4];
  const float* c0        = (const float*)d_in[5];
  const float* emb_table = (const float*)d_in[6];
  const float* w_ih      = (const float*)d_in[7];
  const float* w_hh      = (const float*)d_in[8];
  const float* b_ih      = (const float*)d_in[9];
  const float* b_hh      = (const float*)d_in[10];
  const float* nce_w     = (const float*)d_in[11];
  const float* nce_b     = (const float*)d_in[12];
  const float* noiseprob = (const float*)d_in[13];
  float* out = (float*)d_out;

  char* ws = (char*)d_ws;
  size_t off = 0;
  auto carve = [&](size_t bytes) -> char* {
    char* p = ws + off;
    off += (bytes + 255) & ~(size_t)255;
    return p;
  };
  uint16_t* wihb     = (uint16_t*)carve((size_t)G4H * E_DIM * sizeof(uint16_t));   // 8 MB
  uint16_t* whhb     = (uint16_t*)carve((size_t)G4H * H_DIM * sizeof(uint16_t));   // 8 MB
  uint16_t* emb_bf   = (uint16_t*)carve((size_t)T_LEN * B_SZ * E_DIM * 2);         // 33.5 MB
  uint16_t* hs_bf    = (uint16_t*)carve((size_t)T_LEN * B_SZ * H_DIM * 2);         // 33.5 MB
  uint16_t* h_bf     = (uint16_t*)carve((size_t)B_SZ * H_DIM * 2);                 // 256 KB
  float*    cbuf     = (float*)carve((size_t)B_SZ * H_DIM * 4);                    // 512 KB
  float*    gates    = (float*)carve((size_t)B_SZ * G4H * 4);                      // 2 MB
  float*    bias_sum = (float*)carve((size_t)G4H * 4);
  float*    NLpart   = (float*)carve((size_t)T_LEN * B_SZ * NTILES_NOISE * 4);     // 4.1 MB
  float*    ts       = (float*)carve((size_t)T_LEN * B_SZ * 4);

  // prep
  cvt_weights<<<16384, 256, 0, stream>>>(w_ih, w_hh, wihb, whhb);
  prep_state<<<512, 256, 0, stream>>>(b_ih, b_hh, bias_sum, h0, c0, h_bf, cbuf);
  emb_gather<<<65536, 256, 0, stream>>>(input_ids, emb_table, emb_bf);

  // sequential LSTM (graph-captured chain of 256 kernels)
  for (int t = 0; t < T_LEN; ++t) {
    lstm_gates<<<dim3(8, 8), 256, 0, stream>>>(emb_bf, h_bf, wihb, whhb, bias_sum, gates, t);
    lstm_cell<<<512, 256, 0, stream>>>(gates, cbuf, h_bf, hs_bf, sent_lens, t);
  }

  // NCE
  nce_noise<<<dim3(8, T_LEN), 256, 0, stream>>>(hs_bf, nce_w, nce_b, noiseprob,
                                                noise, NLpart);
  nce_diag<<<2048, 256, 0, stream>>>(hs_bf, nce_w, nce_b, target, ts);
  final_reduce<<<1, 256, 0, stream>>>(ts, NLpart, target, noiseprob, input_ids, out);
}
